// TransformerDecoderLayer_27410481283732
// MI455X (gfx1250) — matrix operand order repaired and checked by reading, not compiled, not run on hardware
//
#include <hip/hip_runtime.h>
#include <hip/hip_bf16.h>

// ---------------------------------------------------------------------------
// Types
// ---------------------------------------------------------------------------
typedef __attribute__((ext_vector_type(16))) __bf16 v16bf;
typedef __attribute__((ext_vector_type(8)))  float  v8f;
typedef int v4i __attribute__((vector_size(16)));

#define D_MODEL 1024
#define N_HEADS 16
#define D_HEAD  64
#define F_FF    4096
#define BATCH   8
#define SEQ     1024
#define LN_EPS  1e-6f

#define LDS_PAD 40   // padded row stride (elements) for LDS tiles: 80B, 16B-aligned

// ---------------------------------------------------------------------------
// CDNA5 async-to-LDS copy (16B per lane), with safe fallback
// ---------------------------------------------------------------------------
__device__ inline void async_copy16(const __bf16* g, __bf16* l) {
#if __has_builtin(__builtin_amdgcn_global_load_async_to_lds_b128)
    __builtin_amdgcn_global_load_async_to_lds_b128(
        (__attribute__((address_space(1))) v4i*)g,
        (__attribute__((address_space(3))) v4i*)l, 0, 0);
#else
    *(uint4*)l = *(const uint4*)g;
#endif
}

__device__ inline void wait_async() {
#if __has_builtin(__builtin_amdgcn_s_wait_asynccnt)
    __builtin_amdgcn_s_wait_asynccnt(0);
#else
    asm volatile("s_wait_asynccnt 0x0" ::: "memory");
#endif
}

// ---------------------------------------------------------------------------
// WMMA fragment loaders (wave32; A and B both use contiguous k = 16*(lane>>4) + j)
// ---------------------------------------------------------------------------

// A-matrix 16x32 bf16 from row-major A[m][k], leading dim lda.
// lane L: m = m0 + (L&15); k-base = k0 + (L>>4)*16; element j -> k += j
__device__ inline v16bf load_a_frag(const __bf16* __restrict__ A, int lda, int m0, int k0) {
    int lane = threadIdx.x & 31;
    const __bf16* p = A + (size_t)(m0 + (lane & 15)) * lda + (k0 + ((lane >> 4) << 4));
    v16bf a;
#pragma unroll
    for (int j = 0; j < 16; ++j) {
        int k = j;
        a[j] = p[k];
    }
    return a;
}

// B-matrix 32x16 from a TRANSPOSED store BT[n][k] with leading dim ld.
// lane L: n = n0 + (L&15); element j -> k = (L>>4)*16 + j  (16 contiguous elements)
__device__ inline v16bf load_bt_frag(const __bf16* __restrict__ BT, int ld, int n0) {
    int lane = threadIdx.x & 31;
    const __bf16* p = BT + (size_t)(n0 + (lane & 15)) * ld + ((lane >> 4) << 4);
    v16bf b;
#pragma unroll
    for (int j = 0; j < 16; ++j) b[j] = p[j];
    return b;
}

// B-matrix 32x16 where B[k][n] = Krows[n][k] (K^T), Krows row-major with ld.
__device__ inline v16bf load_bT_frag(const __bf16* __restrict__ Krows, int ld, int k0, int n0) {
    int lane = threadIdx.x & 31;
    const __bf16* p = Krows + (size_t)(n0 + (lane & 15)) * ld + (k0 + ((lane >> 4) << 4));
    v16bf b;
#pragma unroll
    for (int j = 0; j < 16; ++j) b[j] = p[j];
    return b;
}

// B-matrix 32x16 for PV from V^T stored [D, TOT]: n = d, k = key.
__device__ inline v16bf load_vT_frag(const __bf16* __restrict__ VT, size_t ldv,
                                     int d0, size_t key0) {
    int lane = threadIdx.x & 31;
    const __bf16* p = VT + (size_t)(d0 + (lane & 15)) * ldv + key0 + ((lane >> 4) << 4);
    v16bf b;
#pragma unroll
    for (int j = 0; j < 16; ++j) b[j] = p[j];
    return b;
}

// ---------------------------------------------------------------------------
// Elementwise: f32 -> bf16 convert
// ---------------------------------------------------------------------------
__global__ void k_f32_to_bf16(const float* __restrict__ in, __bf16* __restrict__ out, int n) {
    int i = blockIdx.x * blockDim.x + threadIdx.x;
    if (i < n) out[i] = (__bf16)in[i];
}

// ---------------------------------------------------------------------------
// LayerNorm: one block per row of D_MODEL floats, bf16 output
// ---------------------------------------------------------------------------
__global__ void k_layernorm(const float* __restrict__ X, const float* __restrict__ g,
                            const float* __restrict__ bta, __bf16* __restrict__ Y) {
    __shared__ float rs[256], rss[256];
    int row = blockIdx.x;
    const float* x = X + (size_t)row * D_MODEL;
    float s = 0.f, ss = 0.f;
    for (int i = threadIdx.x; i < D_MODEL; i += 256) {
        float v = x[i];
        s += v; ss += v * v;
    }
    rs[threadIdx.x] = s; rss[threadIdx.x] = ss;
    __syncthreads();
    for (int off = 128; off > 0; off >>= 1) {
        if (threadIdx.x < off) {
            rs[threadIdx.x]  += rs[threadIdx.x + off];
            rss[threadIdx.x] += rss[threadIdx.x + off];
        }
        __syncthreads();
    }
    float mu  = rs[0] * (1.f / D_MODEL);
    float var = rss[0] * (1.f / D_MODEL) - mu * mu;
    var = fmaxf(var, 0.f);
    float inv = 1.f / (sqrtf(var) + LN_EPS);
    __bf16* y = Y + (size_t)row * D_MODEL;
    for (int i = threadIdx.x; i < D_MODEL; i += 256)
        y[i] = (__bf16)((x[i] - mu) * inv * g[i] + bta[i]);
}

// ---------------------------------------------------------------------------
// Blocked bf16 WMMA GEMM with LDS double buffering.
//   C[M,N] = A[M,K] @ W[K,N] + bias (+res) (ReLU?)
// Block: 256 threads = 8 waves (4 along M x 2 along N), block tile 64x128.
// A staged via async-to-LDS; B staged transposed for contiguous ds_load_b128.
// ---------------------------------------------------------------------------
__global__ void k_gemm(const __bf16* __restrict__ A, const __bf16* __restrict__ W,
                       const float* __restrict__ bias, const float* __restrict__ res,
                       float* __restrict__ Cf, __bf16* __restrict__ Cb,
                       __bf16* __restrict__ CbT,
                       int M, int N, int K, int relu) {
    __shared__ __align__(16) __bf16 Alds[2][64 * LDS_PAD];
    __shared__ __align__(16) __bf16 Blds[2][128 * LDS_PAD];

    int nb = N >> 7;                       // blocks along N (BN=128)
    int tM = (int)(blockIdx.x / nb) << 6;  // BM=64
    int tN = (int)(blockIdx.x % nb) << 7;

    int wave = threadIdx.x >> 5;
    int mw = wave >> 1;                    // 0..3
    int nw = wave & 1;                     // 0..1
    int lane = threadIdx.x & 31;

    // --- tile staging ---
    int rowA = threadIdx.x >> 2;           // 0..63
    int c8   = (threadIdx.x & 3) << 3;     // 0,8,16,24
    int kr   = threadIdx.x >> 3;           // 0..31
    int nc0  = (threadIdx.x & 7) << 4;     // 0..112 step 16

    auto stage = [&](int buf, int kt) {
        // A tile 64x32, 16B async copy per thread
        async_copy16(A + (size_t)(tM + rowA) * K + kt + c8,
                     &Alds[buf][rowA * LDS_PAD + c8]);
        // B tile 32x128, stored transposed: Blds[n][k]
        const __bf16* g = W + (size_t)(kt + kr) * N + tN + nc0;
#pragma unroll
        for (int i = 0; i < 16; ++i)
            Blds[buf][(nc0 + i) * LDS_PAD + kr] = g[i];
    };

    v8f c0 = {}, c1 = {}, c2 = {}, c3 = {};

    int nk = K >> 5;
    stage(0, 0);
    wait_async();
    __syncthreads();

    for (int i = 0; i < nk; ++i) {
        int cur = i & 1;
        if (i + 1 < nk) stage(cur ^ 1, (i + 1) << 5);

        v16bf a = load_a_frag(&Alds[cur][0], LDS_PAD, mw << 4, 0);
        {
            v16bf b = load_bt_frag(&Blds[cur][0], LDS_PAD, (nw << 6) + 0);
            c0 = __builtin_amdgcn_wmma_f32_16x16x32_bf16(false, a, false, b, (short)0, c0, false, false);
        }
        {
            v16bf b = load_bt_frag(&Blds[cur][0], LDS_PAD, (nw << 6) + 16);
            c1 = __builtin_amdgcn_wmma_f32_16x16x32_bf16(false, a, false, b, (short)0, c1, false, false);
        }
        {
            v16bf b = load_bt_frag(&Blds[cur][0], LDS_PAD, (nw << 6) + 32);
            c2 = __builtin_amdgcn_wmma_f32_16x16x32_bf16(false, a, false, b, (short)0, c2, false, false);
        }
        {
            v16bf b = load_bt_frag(&Blds[cur][0], LDS_PAD, (nw << 6) + 48);
            c3 = __builtin_amdgcn_wmma_f32_16x16x32_bf16(false, a, false, b, (short)0, c3, false, false);
        }

        wait_async();
        __syncthreads();
    }

    // --- epilogue ---
    int half = lane >> 4;
    int lc   = lane & 15;
    int rbase = tM + (mw << 4) + (half << 3);
    v8f accs[4] = { c0, c1, c2, c3 };
#pragma unroll
    for (int sub = 0; sub < 4; ++sub) {
        int col = tN + (nw << 6) + (sub << 4) + lc;
        float bv = bias[col];
#pragma unroll
        for (int r = 0; r < 8; ++r) {
            int row = rbase + r;
            size_t idx = (size_t)row * N + col;
            float v = accs[sub][r] + bv;
            if (res)  v += res[idx];
            if (relu) v = fmaxf(v, 0.f);
            if (Cf)  Cf[idx] = v;
            if (Cb)  Cb[idx] = (__bf16)v;
            if (CbT) CbT[(size_t)col * M + row] = (__bf16)v;
        }
    }
}

// ---------------------------------------------------------------------------
// Flash attention, one wave per (b, h, 16-row q tile), 32 keys per step.
// Q,K in [B, S, D_MODEL] bf16; V passed TRANSPOSED as VT[D_MODEL][B*Sk] bf16.
// O written bf16 in [B, S, D_MODEL].
// ---------------------------------------------------------------------------
__global__ void k_attn(const __bf16* __restrict__ Q, const __bf16* __restrict__ K,
                       const __bf16* __restrict__ VT, __bf16* __restrict__ O,
                       int Sq, int Sk, int causal) {
    __shared__ __align__(16) __bf16 pbuf[8][16 * 32];

    int wslot = threadIdx.x >> 5;
    int wid   = blockIdx.x * (blockDim.x >> 5) + wslot;
    int qtiles = Sq >> 4;
    int qt = wid % qtiles;
    int h  = (wid / qtiles) % N_HEADS;
    int b  = wid / (qtiles * N_HEADS);

    int lane = threadIdx.x & 31;
    int half = lane >> 4;
    int lc   = lane & 15;

    const __bf16* Qp = Q + (size_t)b * Sq * D_MODEL + h * D_HEAD;
    const __bf16* Kp = K + (size_t)b * Sk * D_MODEL + h * D_HEAD;
    size_t ldv = (size_t)BATCH * Sk;           // rows of VT
    size_t vkey0 = (size_t)b * Sk;

    int q0 = qt << 4;

    v16bf qa0 = load_a_frag(Qp, D_MODEL, q0, 0);
    v16bf qa1 = load_a_frag(Qp, D_MODEL, q0, 32);

    float run_m[8], run_l[8];
#pragma unroll
    for (int r = 0; r < 8; ++r) { run_m[r] = -1e30f; run_l[r] = 0.f; }
    v8f o0 = {}, o1 = {}, o2 = {}, o3 = {};

    int kend = causal ? ((q0 + 16 < Sk) ? (q0 + 16) : Sk) : Sk;
    const float scale = 0.125f;  // 1/sqrt(64)

    for (int kt = 0; kt < kend; kt += 32) {
        v8f s0 = {}, s1 = {};
        {
            v16bf b0 = load_bT_frag(Kp, D_MODEL, 0,  kt);
            s0 = __builtin_amdgcn_wmma_f32_16x16x32_bf16(false, qa0, false, b0, (short)0, s0, false, false);
            v16bf b1 = load_bT_frag(Kp, D_MODEL, 32, kt);
            s0 = __builtin_amdgcn_wmma_f32_16x16x32_bf16(false, qa1, false, b1, (short)0, s0, false, false);
        }
        {
            v16bf b0 = load_bT_frag(Kp, D_MODEL, 0,  kt + 16);
            s1 = __builtin_amdgcn_wmma_f32_16x16x32_bf16(false, qa0, false, b0, (short)0, s1, false, false);
            v16bf b1 = load_bT_frag(Kp, D_MODEL, 32, kt + 16);
            s1 = __builtin_amdgcn_wmma_f32_16x16x32_bf16(false, qa1, false, b1, (short)0, s1, false, false);
        }

        // online softmax; rows live in vgpr index r (+8*half), cols in half's lanes
#pragma unroll
        for (int r = 0; r < 8; ++r) {
            int row = q0 + r + (half << 3);
            float v0 = s0[r] * scale;
            float v1 = s1[r] * scale;
            if (causal) {
                if (kt + lc > row)      v0 = -1e30f;
                if (kt + 16 + lc > row) v1 = -1e30f;
            }
            float mx = fmaxf(v0, v1);
#pragma unroll
            for (int d = 1; d < 16; d <<= 1) mx = fmaxf(mx, __shfl_xor(mx, d, 32));
            float m_new = fmaxf(run_m[r], mx);
            float alpha = __expf(run_m[r] - m_new);
            float p0 = __expf(v0 - m_new);
            float p1 = __expf(v1 - m_new);
            float ls = p0 + p1;
#pragma unroll
            for (int d = 1; d < 16; d <<= 1) ls += __shfl_xor(ls, d, 32);
            run_l[r] = run_l[r] * alpha + ls;
            run_m[r] = m_new;
            o0[r] *= alpha; o1[r] *= alpha; o2[r] *= alpha; o3[r] *= alpha;

            int prow = r + (half << 3);
            pbuf[wslot][prow * 32 + lc]      = (__bf16)p0;
            pbuf[wslot][prow * 32 + 16 + lc] = (__bf16)p1;
        }
        asm volatile("s_wait_dscnt 0x0" ::: "memory");

        v16bf pa = load_a_frag(&pbuf[wslot][0], 32, 0, 0);
        asm volatile("s_wait_dscnt 0x0" ::: "memory");

        {
            v16bf vb = load_vT_frag(VT, ldv, h * D_HEAD + 0,  vkey0 + kt);
            o0 = __builtin_amdgcn_wmma_f32_16x16x32_bf16(false, pa, false, vb, (short)0, o0, false, false);
        }
        {
            v16bf vb = load_vT_frag(VT, ldv, h * D_HEAD + 16, vkey0 + kt);
            o1 = __builtin_amdgcn_wmma_f32_16x16x32_bf16(false, pa, false, vb, (short)0, o1, false, false);
        }
        {
            v16bf vb = load_vT_frag(VT, ldv, h * D_HEAD + 32, vkey0 + kt);
            o2 = __builtin_amdgcn_wmma_f32_16x16x32_bf16(false, pa, false, vb, (short)0, o2, false, false);
        }
        {
            v16bf vb = load_vT_frag(VT, ldv, h * D_HEAD + 48, vkey0 + kt);
            o3 = __builtin_amdgcn_wmma_f32_16x16x32_bf16(false, pa, false, vb, (short)0, o3, false, false);
        }
    }

    __bf16* Op = O + (size_t)b * Sq * D_MODEL + h * D_HEAD;
#pragma unroll
    for (int r = 0; r < 8; ++r) {
        int row = q0 + r + (half << 3);
        float inv = 1.f / run_l[r];
        size_t base = (size_t)row * D_MODEL + lc;
        Op[base +  0] = (__bf16)(o0[r] * inv);
        Op[base + 16] = (__bf16)(o1[r] * inv);
        Op[base + 32] = (__bf16)(o2[r] * inv);
        Op[base + 48] = (__bf16)(o3[r] * inv);
    }
}

// ---------------------------------------------------------------------------
// Host-side orchestration
// ---------------------------------------------------------------------------
extern "C" void kernel_launch(void* const* d_in, const int* in_sizes, int n_in,
                              void* d_out, int out_size, void* d_ws, size_t ws_size,
                              hipStream_t stream) {
    (void)in_sizes; (void)n_in; (void)out_size; (void)ws_size;

    const int ROWS = BATCH * SEQ;           // 8192
    const int ND   = ROWS * D_MODEL;        // 8,388,608

    const float* x     = (const float*)d_in[0];
    const float* src_x = (const float*)d_in[2];

    const float* sa_q_w = (const float*)d_in[4];  const float* sa_q_b = (const float*)d_in[5];
    const float* sa_k_w = (const float*)d_in[6];  const float* sa_k_b = (const float*)d_in[7];
    const float* sa_v_w = (const float*)d_in[8];  const float* sa_v_b = (const float*)d_in[9];
    const float* sa_o_w = (const float*)d_in[10]; const float* sa_o_b = (const float*)d_in[11];
    const float* ca_q_w = (const float*)d_in[12]; const float* ca_q_b = (const float*)d_in[13];
    const float* ca_k_w = (const float*)d_in[14]; const float* ca_k_b = (const float*)d_in[15];
    const float* ca_v_w = (const float*)d_in[16]; const float* ca_v_b = (const float*)d_in[17];
    const float* ca_o_w = (const float*)d_in[18]; const float* ca_o_b = (const float*)d_in[19];
    const float* ff_w1  = (const float*)d_in[20]; const float* ff_b1  = (const float*)d_in[21];
    const float* ff_w2  = (const float*)d_in[22]; const float* ff_b2  = (const float*)d_in[23];
    const float* ln1_g  = (const float*)d_in[24]; const float* ln1_b  = (const float*)d_in[25];
    const float* ln2_g  = (const float*)d_in[26]; const float* ln2_b  = (const float*)d_in[27];
    const float* ln3_g  = (const float*)d_in[28]; const float* ln3_b  = (const float*)d_in[29];

    float* out = (float*)d_out;

    char* wsp = (char*)d_ws;
    auto alloc = [&](size_t bytes) -> void* {
        void* p = wsp;
        wsp += (bytes + 255) & ~(size_t)255;
        return p;
    };

    __bf16* wqkvo[8];
    const float* wsrc[8] = { sa_q_w, sa_k_w, sa_v_w, sa_o_w, ca_q_w, ca_k_w, ca_v_w, ca_o_w };
    for (int i = 0; i < 8; ++i) wqkvo[i] = (__bf16*)alloc((size_t)D_MODEL * D_MODEL * 2);
    __bf16* w1b   = (__bf16*)alloc((size_t)D_MODEL * F_FF * 2);
    __bf16* w2b   = (__bf16*)alloc((size_t)F_FF * D_MODEL * 2);
    __bf16* srcb  = (__bf16*)alloc((size_t)ND * 2);
    __bf16* q1b   = (__bf16*)alloc((size_t)ND * 2);
    __bf16* Qb    = (__bf16*)alloc((size_t)ND * 2);
    __bf16* Kb    = (__bf16*)alloc((size_t)ND * 2);
    __bf16* VbT   = (__bf16*)alloc((size_t)ND * 2);   // transposed [D, B*S]
    __bf16* attnb = (__bf16*)alloc((size_t)ND * 2);
    float*  x1f   = (float*) alloc((size_t)ND * 4);
    __bf16* q2b   = (__bf16*)alloc((size_t)ND * 2);
    __bf16* Qcb   = (__bf16*)alloc((size_t)ND * 2);
    __bf16* Kcb   = (__bf16*)alloc((size_t)ND * 2);
    __bf16* VcbT  = (__bf16*)alloc((size_t)ND * 2);   // transposed [D, B*S]
    __bf16* att2b = (__bf16*)alloc((size_t)ND * 2);
    float*  x2f   = (float*) alloc((size_t)ND * 4);
    __bf16* hb    = (__bf16*)alloc((size_t)ND * 2);
    __bf16* ff1b  = (__bf16*)alloc((size_t)ROWS * F_FF * 2);

    auto cvt = [&](const float* src, __bf16* dst, int n) {
        k_f32_to_bf16<<<(n + 255) / 256, 256, 0, stream>>>(src, dst, n);
    };
    auto gemm = [&](const __bf16* A, const __bf16* W, const float* bias, const float* res,
                    float* Cf, __bf16* Cb, __bf16* CbT, int M, int N, int K, int relu) {
        int blocks = (M / 64) * (N / 128);
        k_gemm<<<blocks, 256, 0, stream>>>(A, W, bias, res, Cf, Cb, CbT, M, N, K, relu);
    };

    // 0) precision conversion of weights + src_x
    for (int i = 0; i < 8; ++i) cvt(wsrc[i], wqkvo[i], D_MODEL * D_MODEL);
    cvt(ff_w1, w1b, D_MODEL * F_FF);
    cvt(ff_w2, w2b, F_FF * D_MODEL);
    cvt(src_x, srcb, ND);

    // 1) self-attention block
    k_layernorm<<<ROWS, 256, 0, stream>>>(x, ln1_g, ln1_b, q1b);
    gemm(q1b, wqkvo[0], sa_q_b, nullptr, nullptr, Qb,  nullptr, ROWS, D_MODEL, D_MODEL, 0);
    gemm(q1b, wqkvo[1], sa_k_b, nullptr, nullptr, Kb,  nullptr, ROWS, D_MODEL, D_MODEL, 0);
    gemm(q1b, wqkvo[2], sa_v_b, nullptr, nullptr, nullptr, VbT, ROWS, D_MODEL, D_MODEL, 0);
    {
        int waves = BATCH * N_HEADS * (SEQ / 16);   // 8192
        k_attn<<<waves / 8, 256, 0, stream>>>(Qb, Kb, VbT, attnb, SEQ, SEQ, 1);
    }
    gemm(attnb, wqkvo[3], sa_o_b, x, x1f, nullptr, nullptr, ROWS, D_MODEL, D_MODEL, 0);

    // 2) cross-attention block (src_mask all-false)
    k_layernorm<<<ROWS, 256, 0, stream>>>(x1f, ln2_g, ln2_b, q2b);
    gemm(q2b,  wqkvo[4], ca_q_b, nullptr, nullptr, Qcb, nullptr, ROWS, D_MODEL, D_MODEL, 0);
    gemm(srcb, wqkvo[5], ca_k_b, nullptr, nullptr, Kcb, nullptr, ROWS, D_MODEL, D_MODEL, 0);
    gemm(srcb, wqkvo[6], ca_v_b, nullptr, nullptr, nullptr, VcbT, ROWS, D_MODEL, D_MODEL, 0);
    {
        int waves = BATCH * N_HEADS * (SEQ / 16);
        k_attn<<<waves / 8, 256, 0, stream>>>(Qcb, Kcb, VcbT, att2b, SEQ, SEQ, 0);
    }
    gemm(att2b, wqkvo[7], ca_o_b, x1f, x2f, nullptr, nullptr, ROWS, D_MODEL, D_MODEL, 0);

    // 3) FFN block
    k_layernorm<<<ROWS, 256, 0, stream>>>(x2f, ln3_g, ln3_b, hb);
    gemm(hb,   w1b, ff_b1, nullptr, nullptr, ff1b, nullptr, ROWS, F_FF, D_MODEL, 1);
    gemm(ff1b, w2b, ff_b2, x2f, out, nullptr, nullptr, ROWS, D_MODEL, F_FF, 0);
}
